// GCN_72138270704343
// MI455X (gfx1250) — matrix operand order repaired and checked by reading, not compiled, not run on hardware
//
#include <hip/hip_runtime.h>

#define NNODES 100000
#define NEDGES 1600000
#define HDIM   128
#define CDIM   40
#define CPAD   48

typedef __attribute__((ext_vector_type(16))) __bf16 v16bf;
typedef __attribute__((ext_vector_type(8)))  __bf16 v8bf;
typedef __attribute__((ext_vector_type(8)))  float  v8f;

static __device__ __forceinline__ __bf16 f2bf(float f) {
  union { float f; unsigned u; } cv; cv.f = f;
  unsigned u = cv.u;
  u += 0x7fffu + ((u >> 16) & 1u);            // round-to-nearest-even
  unsigned short h = (unsigned short)(u >> 16);
  union { unsigned short s; __bf16 b; } cb; cb.s = h;
  return cb.b;
}

// ---------------------------------------------------------------- utilities
__global__ void gcn_fill_zero(float* __restrict__ p, long long n) {
  long long i = (long long)blockIdx.x * blockDim.x + threadIdx.x;
  long long stride = (long long)gridDim.x * blockDim.x;
  for (; i < n; i += stride) p[i] = 0.0f;
}

__global__ void gcn_deg_count(const int* __restrict__ src, const int* __restrict__ dst,
                              float* __restrict__ cnt_out, float* __restrict__ cnt_in, int E) {
  int e = blockIdx.x * blockDim.x + threadIdx.x;
  if (e < E) {
    atomicAdd(&cnt_out[src[e]], 1.0f);
    atomicAdd(&cnt_in[dst[e]], 1.0f);
  }
}

__global__ void gcn_to_rsqrt(float* __restrict__ p, int n) {
  int i = blockIdx.x * blockDim.x + threadIdx.x;
  if (i < n) p[i] = rsqrtf(fmaxf(p[i], 1.0f));
}

// W (128 x wcols, row-major fp32) -> Wt (colspad x 128, bf16, zero-padded cols)
__global__ void gcn_wt_convert(const float* __restrict__ W, int wcols, int colspad,
                               __bf16* __restrict__ Wt) {
  int i = blockIdx.x * blockDim.x + threadIdx.x;   // colspad * 128 elements
  if (i >= colspad * HDIM) return;
  int c = i >> 7;          // column of W
  int k = i & 127;         // row of W
  float v = (c < wcols) ? W[(long long)k * wcols + c] : 0.0f;
  Wt[(long long)c * HDIM + k] = f2bf(v);
}

// layer-0 input: Xbf[n,k] = bf16(feat[n,k] * sout[n])
__global__ void gcn_x0_convert(const float* __restrict__ X, const float* __restrict__ sout,
                               __bf16* __restrict__ Xbf, long long total) {
  long long i = (long long)blockIdx.x * blockDim.x + threadIdx.x;
  if (i >= total) return;
  int n = (int)(i >> 7);
  Xbf[i] = f2bf(X[i] * sout[n]);
}

// ---------------------------------------------------------------- WMMA GEMM
// One wave computes a 16 x (NT*16) tile of Y = Xbf @ Wt^T.
// Xbf: M x 128 bf16 row-major (pre-scaled). Wt: cols x 128 bf16 (W transposed).
template <int NT>
__global__ __launch_bounds__(32) void gcn_gemm_wmma(
    const __bf16* __restrict__ Xbf,
    const __bf16* __restrict__ Wt,
    float* __restrict__ Y, int ldy) {
  const int lane = threadIdx.x;            // wave32, one wave per block
  const int m0   = blockIdx.x * 16;
  const int n0   = blockIdx.y * (NT * 16);
  const int half = lane >> 4;              // 0 or 1
  const int l15  = lane & 15;

  const __bf16* xrow = Xbf + (long long)(m0 + l15) * HDIM;

  v8f acc[NT];
#pragma unroll
  for (int t = 0; t < NT; ++t) acc[t] = (v8f){};

#pragma unroll
  for (int kt = 0; kt < 4; ++kt) {         // K = 128 = 4 x 32
    const int kbase = kt * 32;
    // A 16x32 bf16: lane row = l15; contiguous runs [8*half, +8) and [16+8*half, +8)
    v8bf alo = *(const v8bf*)(xrow + kbase + 8 * half);
    v8bf ahi = *(const v8bf*)(xrow + kbase + 16 + 8 * half);
    v16bf a;
#pragma unroll
    for (int i = 0; i < 8; ++i) { a[i] = alo[i]; a[i + 8] = ahi[i]; }

#pragma unroll
    for (int t = 0; t < NT; ++t) {
      // B 32x16 bf16: lane col = n0 + t*16 + l15; runs [8*half, +8) and [16+8*half, +8)
      const __bf16* wcol = Wt + (long long)(n0 + t * 16 + l15) * HDIM + kbase + 8 * half;
      v8bf blo = *(const v8bf*)(wcol);
      v8bf bhi = *(const v8bf*)(wcol + 16);
      v16bf b;
#pragma unroll
      for (int i = 0; i < 8; ++i) { b[i] = blo[i]; b[i + 8] = bhi[i]; }
      acc[t] = __builtin_amdgcn_wmma_f32_16x16x32_bf16(
          false, a, false, b, (short)0, acc[t], false, false);
    }
  }
  // D 16x16 f32: VGPR v -> row M = v + 8*half, col = l15
#pragma unroll
  for (int t = 0; t < NT; ++t)
#pragma unroll
    for (int v = 0; v < 8; ++v)
      Y[(long long)(m0 + v + 8 * half) * ldy + n0 + t * 16 + l15] = acc[t][v];
}

// ---------------------------------------------------------------- edge scatter
// one wave per edge; lane covers 4 contiguous floats -> 512B/row coalesced
__global__ void gcn_scatter(const float* __restrict__ T, const int* __restrict__ src,
                            const int* __restrict__ dst, float* __restrict__ A,
                            int E, int ncols) {
  int wave = blockIdx.x * (blockDim.x >> 5) + (threadIdx.x >> 5);
  int lane = threadIdx.x & 31;
  if (wave >= E) return;
  int s = src[wave], d = dst[wave];
  int c = lane * 4;
  if (c < ncols) {
    const float4 v = *(const float4*)(T + (long long)s * HDIM + c);
    float* o = A + (long long)d * HDIM + c;
    atomicAdd(o + 0, v.x);
    atomicAdd(o + 1, v.y);
    atomicAdd(o + 2, v.z);
    atomicAdd(o + 3, v.w);
  }
}

// ---------------------------------------------------------------- epilogues
// Xbf[n,c] = bf16( relu(agg[n,c]*sin[n] + b[c]) * sout[n] )  (fused next-layer prescale)
__global__ void gcn_finish_bf16(const float* __restrict__ A, const float* __restrict__ sin_,
                                const float* __restrict__ sout, const float* __restrict__ bias,
                                __bf16* __restrict__ Xbf, long long total) {
  long long i = (long long)blockIdx.x * blockDim.x + threadIdx.x;
  if (i >= total) return;
  int n = (int)(i >> 7);
  int c = (int)(i & 127);
  float v = fmaxf(A[i] * sin_[n] + bias[c], 0.0f) * sout[n];
  Xbf[i] = f2bf(v);
}

__global__ void gcn_final_out(const float* __restrict__ A, const float* __restrict__ sin_,
                              const float* __restrict__ b2, float* __restrict__ out,
                              long long total) {
  long long i = (long long)blockIdx.x * blockDim.x + threadIdx.x;
  if (i >= total) return;
  int n = (int)(i / CDIM);
  int c = (int)(i % CDIM);
  out[i] = A[(long long)n * HDIM + c] * sin_[n] + b2[c];
}

// ---------------------------------------------------------------- launcher
extern "C" void kernel_launch(void* const* d_in, const int* in_sizes, int n_in,
                              void* d_out, int out_size, void* d_ws, size_t ws_size,
                              hipStream_t stream) {
  (void)in_sizes; (void)n_in; (void)out_size; (void)ws_size;
  const float* feat = (const float*)d_in[0];
  const int*   src  = (const int*)d_in[1];
  const int*   dst  = (const int*)d_in[2];
  const float* W0   = (const float*)d_in[3];
  const float* b0   = (const float*)d_in[4];
  const float* W1   = (const float*)d_in[5];
  const float* b1   = (const float*)d_in[6];
  const float* W2   = (const float*)d_in[7];
  const float* b2   = (const float*)d_in[8];
  float* out = (float*)d_out;

  char* w = (char*)d_ws;
  size_t off = 0;
  auto take = [&](size_t bytes) -> void* {
    void* p = w + off;
    off = (off + bytes + 255) & ~(size_t)255;
    return p;
  };
  float*  sout = (float*)take((size_t)NNODES * 4);
  float*  sin_ = (float*)take((size_t)NNODES * 4);
  __bf16* Xbf  = (__bf16*)take((size_t)NNODES * HDIM * 2);  // pre-scaled activations
  float*  bufT = (float*)take((size_t)NNODES * HDIM * 4);   // X@W
  float*  bufA = (float*)take((size_t)NNODES * HDIM * 4);   // aggregated
  __bf16* W0t  = (__bf16*)take((size_t)HDIM * HDIM * 2);
  __bf16* W1t  = (__bf16*)take((size_t)HDIM * HDIM * 2);
  __bf16* W2t  = (__bf16*)take((size_t)CPAD * HDIM * 2);

  const long long tot128 = (long long)NNODES * HDIM;

  // degrees -> rsqrt scales
  gcn_fill_zero<<<1024, 256, 0, stream>>>(sout, NNODES);
  gcn_fill_zero<<<1024, 256, 0, stream>>>(sin_, NNODES);
  gcn_deg_count<<<(NEDGES + 255) / 256, 256, 0, stream>>>(src, dst, sout, sin_, NEDGES);
  gcn_to_rsqrt<<<(NNODES + 255) / 256, 256, 0, stream>>>(sout, NNODES);
  gcn_to_rsqrt<<<(NNODES + 255) / 256, 256, 0, stream>>>(sin_, NNODES);

  // weights -> bf16 transposed
  gcn_wt_convert<<<(HDIM * HDIM + 255) / 256, 256, 0, stream>>>(W0, HDIM, HDIM, W0t);
  gcn_wt_convert<<<(HDIM * HDIM + 255) / 256, 256, 0, stream>>>(W1, HDIM, HDIM, W1t);
  gcn_wt_convert<<<(CPAD * HDIM + 255) / 256, 256, 0, stream>>>(W2, CDIM, CPAD, W2t);

  const dim3 gemmGridH(NNODES / 16, HDIM / 64);    // 6250 x 2, NT=4
  const dim3 gemmGridC(NNODES / 16, 1);            // 6250 x 1, NT=3 (48 cols)
  const unsigned g128 = (unsigned)((tot128 + 255) / 256);

  // ---- layer 0
  gcn_x0_convert<<<g128, 256, 0, stream>>>(feat, sout, Xbf, tot128);
  gcn_gemm_wmma<4><<<gemmGridH, 32, 0, stream>>>(Xbf, W0t, bufT, HDIM);
  gcn_fill_zero<<<4096, 256, 0, stream>>>(bufA, tot128);
  gcn_scatter<<<(NEDGES + 7) / 8, 256, 0, stream>>>(bufT, src, dst, bufA, NEDGES, HDIM);
  gcn_finish_bf16<<<g128, 256, 0, stream>>>(bufA, sin_, sout, b0, Xbf, tot128);

  // ---- layer 1
  gcn_gemm_wmma<4><<<gemmGridH, 32, 0, stream>>>(Xbf, W1t, bufT, HDIM);
  gcn_fill_zero<<<4096, 256, 0, stream>>>(bufA, tot128);
  gcn_scatter<<<(NEDGES + 7) / 8, 256, 0, stream>>>(bufT, src, dst, bufA, NEDGES, HDIM);
  gcn_finish_bf16<<<g128, 256, 0, stream>>>(bufA, sin_, sout, b1, Xbf, tot128);

  // ---- layer 2 (C = 40, padded to 48 cols in scratch)
  gcn_gemm_wmma<3><<<gemmGridC, 32, 0, stream>>>(Xbf, W2t, bufT, HDIM);
  gcn_fill_zero<<<4096, 256, 0, stream>>>(bufA, tot128);
  gcn_scatter<<<(NEDGES + 7) / 8, 256, 0, stream>>>(bufT, src, dst, bufA, NEDGES, CDIM);
  const long long totOut = (long long)NNODES * CDIM;
  gcn_final_out<<<(unsigned)((totOut + 255) / 256), 256, 0, stream>>>(bufA, sin_, b2, out, totOut);
}
